// AdaptiveSpanAttention_49340584297079
// MI455X (gfx1250) — compile-verified, matrix operand order checked by reading, not run on hardware
//
#include <hip/hip_runtime.h>
#include <hip/hip_bf16.h>
#include <math.h>

// ---- CDNA5 WMMA types ----
typedef __attribute__((ext_vector_type(16))) __bf16 bf16x16;
typedef __attribute__((ext_vector_type(8)))  float  f32x8;

#define cB 4
#define cT 1024
#define cD 1024
#define cH 16
#define cDH 64
#define R_SPAN 256.0f

// ---- Tensor Data Mover availability (device pass only; host sees 0) ----
#if __has_builtin(__builtin_amdgcn_tensor_load_to_lds) && \
    __has_builtin(__builtin_amdgcn_s_wait_tensorcnt)
#define USE_TDM 1
#else
#define USE_TDM 0
#endif

#if USE_TDM
typedef __attribute__((ext_vector_type(4))) unsigned int u32x4;
typedef __attribute__((ext_vector_type(4))) int          i32x4;
typedef __attribute__((ext_vector_type(8))) int          i32x8;

// Issue a TDM 2D tile load: rows x cols bf16 elements, row stride in elems.
// All arguments must be wave-uniform. D# packed per CDNA5 ISA section 8.
__device__ __forceinline__ void tdm_load_tile_2d(const __bf16* gsrc,
                                                 unsigned lds_byte_off,
                                                 unsigned rows, unsigned cols,
                                                 unsigned row_stride_elems) {
  const unsigned long long ga = (unsigned long long)(const void*)gsrc;
  u32x4 g0 = {};
  g0[0] = 1u;                                   // count=1, user descriptor
  g0[1] = lds_byte_off;                         // lds_addr [63:32]
  g0[2] = (unsigned)(ga & 0xffffffffull);       // global_addr [95:64]
  g0[3] = (unsigned)((ga >> 32) & 0x1ffffffull) // global_addr [120:96]
          | (2u << 30);                         // type=2 ("image")
  const unsigned td0 = 1u << 20, td1 = 1u << 20;  // huge dims: no OOB clip
  i32x8 g1 = {};
  g1[0] = (int)(1u << 16);                      // data_size=1 (2 bytes)
  g1[1] = (int)((td0 & 0xffffu) << 16);                               // tensor_dim0 lo
  g1[2] = (int)(((td0 >> 16) & 0xffffu) | ((td1 & 0xffffu) << 16));   // td0 hi | td1 lo
  g1[3] = (int)(((td1 >> 16) & 0xffffu) | ((cols & 0xffffu) << 16));  // td1 hi | tile_dim0
  g1[4] = (int)(rows & 0xffffu);                // tile_dim1 (tile_dim2=0)
  g1[5] = (int)row_stride_elems;                // tensor_dim0_stride [47:0]
  g1[6] = 0;
  g1[7] = 0;                                    // tensor_dim1_stride unused (2D)
  i32x4 z4 = {};
#if __clang_major__ >= 23
  i32x8 z8 = {};
  __builtin_amdgcn_tensor_load_to_lds(g0, g1, z4, z4, z8, 0);
#else
  __builtin_amdgcn_tensor_load_to_lds(g0, g1, z4, z4, 0);
#endif
}

__device__ __forceinline__ unsigned lds_offset_of(const void* p) {
  return (unsigned)(uintptr_t)(__attribute__((address_space(3))) const void*)p;
}
#endif  // USE_TDM

// =====================================================================
// f32 -> bf16 conversion (grid-stride)
// =====================================================================
__global__ void cvt_bf16_kernel(const float* __restrict__ src,
                                __bf16* __restrict__ dst, int n) {
  int i = blockIdx.x * blockDim.x + threadIdx.x;
  int stride = gridDim.x * blockDim.x;
  for (; i < n; i += stride) dst[i] = (__bf16)src[i];
}

// =====================================================================
// Generic bf16 WMMA GEMM:  C[M,N] = A[M,K] @ Bw[N,K]^T
//   MODE 0: store C as bf16 row-major (ld = N)
//   MODE 1: store C transposed per-batch for V: Vt[b][e][t]  (bf16)
//   MODE 2: store C as f32 + bias  (final output projection)
// Block = 256 threads = 8 waves, wave tile 32x64, block tile 128x128.
// =====================================================================
template <int MODE>
__global__ void __launch_bounds__(256, 1)
gemm_bf16_kernel(const __bf16* __restrict__ A, const __bf16* __restrict__ Bw,
                 __bf16* __restrict__ Cb, float* __restrict__ Cf,
                 const float* __restrict__ bias, int M, int N, int K) {
  const int lane = threadIdx.x & 31;
  const int wave = threadIdx.x >> 5;
  const int lo = lane & 15, hi = lane >> 4;
  const int m0 = blockIdx.y * 128 + (wave & 3) * 32;   // 4 waves along M
  const int n0 = blockIdx.x * 128 + (wave >> 2) * 64;  // 2 waves along N

  // Loop-invariant per-lane base pointers (fragment-load addresses)
  const __bf16* pa0 = A + (size_t)(m0 + lo) * K + hi * 16;
  const __bf16* pa1 = pa0 + (size_t)16 * K;
  const __bf16* pb0 = Bw + (size_t)(n0 + lo) * K + hi * 16;

  f32x8 acc[2][4] = {};

  for (int k0 = 0; k0 < K; k0 += 32) {
    bf16x16 a0 = *(const bf16x16*)(pa0 + k0);
    bf16x16 a1 = *(const bf16x16*)(pa1 + k0);
    bf16x16 bf[4];
#pragma unroll
    for (int j = 0; j < 4; ++j)
      bf[j] = *(const bf16x16*)(pb0 + (size_t)(j * 16) * K + k0);
#pragma unroll
    for (int j = 0; j < 4; ++j) {
      acc[0][j] = __builtin_amdgcn_wmma_f32_16x16x32_bf16(
          false, a0, false, bf[j], (short)0, acc[0][j], false, false);
      acc[1][j] = __builtin_amdgcn_wmma_f32_16x16x32_bf16(
          false, a1, false, bf[j], (short)0, acc[1][j], false, false);
    }
  }

#pragma unroll
  for (int i = 0; i < 2; ++i) {
#pragma unroll
    for (int j = 0; j < 4; ++j) {
#pragma unroll
      for (int r = 0; r < 8; ++r) {
        const int row = m0 + i * 16 + r + 8 * hi;  // C layout: VGPR r + half
        const int col = n0 + j * 16 + lo;
        const float v = acc[i][j][r];
        if (MODE == 0) {
          Cb[(size_t)row * N + col] = (__bf16)v;
        } else if (MODE == 1) {
          const int b = row >> 10;  // row = b*T + t (T=1024)
          const int t = row & (cT - 1);
          Cb[((size_t)b * cD + col) * cT + t] = (__bf16)v;  // Vt[b][e][t]
        } else {
          Cf[(size_t)row * N + col] = v + bias[col];
        }
      }
    }
  }
}

// =====================================================================
// xmean[b][d] = mean_t x[b,t,d]
// =====================================================================
__global__ void mean_kernel(const float* __restrict__ x,
                            float* __restrict__ xmean) {
  int idx = blockIdx.x * blockDim.x + threadIdx.x;
  if (idx >= cB * cD) return;
  const int b = idx / cD, d = idx % cD;
  const float* p = x + (size_t)b * cT * cD + d;
  float s = 0.f;
  for (int t = 0; t < cT; ++t) s += p[(size_t)t * cD];
  xmean[idx] = s * (1.0f / (float)cT);
}

// =====================================================================
// z[b,h] = T * sigmoid(dot(xmean[b], Wspan[h]) + bspan[h])
// =====================================================================
__global__ void span_kernel(const float* __restrict__ xmean,
                            const float* __restrict__ Wspan,
                            const float* __restrict__ bspan,
                            float* __restrict__ zbuf) {
  const int b = blockIdx.x / cH, h = blockIdx.x % cH;
  __shared__ float red[256];
  float p = 0.f;
  for (int d = threadIdx.x; d < cD; d += 256)
    p += xmean[b * cD + d] * Wspan[h * cD + d];
  red[threadIdx.x] = p;
  __syncthreads();
  for (int s = 128; s > 0; s >>= 1) {
    if ((int)threadIdx.x < s) red[threadIdx.x] += red[threadIdx.x + s];
    __syncthreads();
  }
  if (threadIdx.x == 0) {
    const float logit = red[0] + bspan[h];
    zbuf[blockIdx.x] = (float)cT / (1.0f + __expf(-logit));
  }
}

// =====================================================================
// Flash-style adaptive-span attention (reversed-causal: j >= i).
// grid = (T/128, B*H), block = 256 (8 waves). Each wave: 16 rows, dh=64.
// Per j-step the whole block shares one 64x64 K tile + one 64x64 Vt tile,
// staged into LDS by the Tensor Data Mover (TENSOR_LOAD_TO_LDS +
// s_wait_tensorcnt), fallback = cooperative vector copy.
// =====================================================================
__global__ void __launch_bounds__(256, 1)
attn_kernel(const __bf16* __restrict__ Qb, const __bf16* __restrict__ Kb,
            const __bf16* __restrict__ Vtb, const float* __restrict__ zbuf,
            __bf16* __restrict__ Ob) {
  __shared__ __bf16 Pl[8][16 * 64];  // per-wave 16x64 prob tile (16 KB)
  __shared__ __bf16 Ks[64 * 64];     // K tile   (8 KB)
  __shared__ __bf16 Vs[64 * 64];     // Vt tile  (8 KB)

  const int lane = threadIdx.x & 31, wave = threadIdx.x >> 5;
  const int lo = lane & 15, hi = lane >> 4;
  const int bh = blockIdx.y;
  const int b = bh / cH, h = bh % cH;
  const int r0 = blockIdx.x * 128 + wave * 16;  // this wave's 16 rows

  const __bf16* Qp = Qb + ((size_t)b * cT) * cD + h * cDH;
  const __bf16* Kp = Kb + ((size_t)b * cT) * cD + h * cDH;
  const __bf16* Vp = Vtb + ((size_t)b * cD + h * cDH) * cT;  // Vt[d][t]
  const float z = zbuf[bh];

  // Q fragments for both K-halves of dh=64 (resident for whole j loop)
  bf16x16 qa[2];
#pragma unroll
  for (int kk = 0; kk < 2; ++kk)
    qa[kk] = *(const bf16x16*)(Qp + (size_t)(r0 + lo) * cD + kk * 32 + hi * 16);

  f32x8 acc[4] = {};
  float m[8], l[8];
#pragma unroll
  for (int r = 0; r < 8; ++r) { m[r] = -INFINITY; l[r] = 0.f; }

  // Block-uniform loop bounds (z is per (b,h) => uniform) so barriers are safe
  const int jstart = blockIdx.x * 128;
  const int jend = (int)fminf((float)cT,
                              (float)(blockIdx.x * 128 + 127) + z + R_SPAN + 2.0f);

  for (int jt = jstart; jt < jend; jt += 64) {
    __syncthreads();  // previous iteration's LDS reads complete
    // ---- stage K/Vt 64x64 tiles into LDS ----
#if USE_TDM
    if (wave == 0) {
      tdm_load_tile_2d(Kp + (size_t)jt * cD, lds_offset_of(&Ks[0]), 64, 64, cD);
      tdm_load_tile_2d(Vp + jt, lds_offset_of(&Vs[0]), 64, 64, cT);
      __builtin_amdgcn_s_wait_tensorcnt(0);
    }
#else
    {
      const int t = threadIdx.x;
      const int rr = t >> 2;          // 0..63
      const int cc = (t & 3) * 16;    // 0,16,32,48
      *(bf16x16*)(&Ks[rr * 64 + cc]) =
          *(const bf16x16*)(Kp + (size_t)(jt + rr) * cD + cc);
      *(bf16x16*)(&Vs[rr * 64 + cc]) =
          *(const bf16x16*)(Vp + (size_t)rr * cT + jt + cc);
    }
#endif
    __syncthreads();  // tiles visible to all waves

    // --- S tile: 16 rows x 64 cols over dh=64 (K fragments from LDS) ---
    f32x8 s4[4] = {};
#pragma unroll
    for (int kk = 0; kk < 2; ++kk) {
#pragma unroll
      for (int jn = 0; jn < 4; ++jn) {
        bf16x16 kb = *(const bf16x16*)(&Ks[(jn * 16 + lo) * 64 + kk * 32 + hi * 16]);
        s4[jn] = __builtin_amdgcn_wmma_f32_16x16x32_bf16(
            false, qa[kk], false, kb, (short)0, s4[jn], false, false);
      }
    }

    // --- online softmax + span mask; build P tile in per-wave LDS ---
#pragma unroll
    for (int r = 0; r < 8; ++r) {
      const int row = r0 + r + 8 * hi;
      float sv[4];
      float mx = -INFINITY;
#pragma unroll
      for (int jn = 0; jn < 4; ++jn) {
        const int j = jt + jn * 16 + lo;
        float s = s4[jn][r] * 0.125f;     // / sqrt(dh)
        s = (j >= row) ? s : -INFINITY;   // reversed-causal mask
        sv[jn] = s;
        mx = fmaxf(mx, s);
      }
#pragma unroll
      for (int off = 1; off < 16; off <<= 1)
        mx = fmaxf(mx, __shfl_xor(mx, off, 32));  // 16-lane row max
      const float mn = fmaxf(m[r], mx);
      const float alpha = (mn == -INFINITY) ? 1.0f : __expf(m[r] - mn);
      m[r] = mn;
      float rs = 0.f;
#pragma unroll
      for (int jn = 0; jn < 4; ++jn) {
        const int j = jt + jn * 16 + lo;
        const float dist = (float)(j - row);
        const float chi =
            fminf(fmaxf((R_SPAN + z - dist) * (1.0f / R_SPAN), 0.0f), 1.0f);
        const float ww =
            (sv[jn] == -INFINITY) ? 0.0f : __expf(sv[jn] - mn) * chi;
        rs += ww;
        Pl[wave][(r + 8 * hi) * 64 + jn * 16 + lo] = (__bf16)ww;
      }
#pragma unroll
      for (int off = 1; off < 16; off <<= 1)
        rs += __shfl_xor(rs, off, 32);            // 16-lane row sum
      l[r] = l[r] * alpha + rs;
#pragma unroll
      for (int dn = 0; dn < 4; ++dn) acc[dn][r] *= alpha;
    }

    // --- acc += P(16x64) @ V(64x64): both operands from LDS ---
#pragma unroll
    for (int kk = 0; kk < 2; ++kk) {
      bf16x16 pa = *(const bf16x16*)(&Pl[wave][lo * 64 + kk * 32 + hi * 16]);
#pragma unroll
      for (int dn = 0; dn < 4; ++dn) {
        bf16x16 vb = *(const bf16x16*)(&Vs[(dn * 16 + lo) * 64 + kk * 32 + hi * 16]);
        acc[dn] = __builtin_amdgcn_wmma_f32_16x16x32_bf16(
            false, pa, false, vb, (short)0, acc[dn], false, false);
      }
    }
  }

  // --- normalize and store (bf16, row-major (B*T, D)) ---
#pragma unroll
  for (int dn = 0; dn < 4; ++dn) {
#pragma unroll
    for (int r = 0; r < 8; ++r) {
      const int row = r0 + r + 8 * hi;
      const float v = acc[dn][r] / (l[r] + 1e-8f);
      Ob[((size_t)b * cT + row) * cD + h * cDH + dn * 16 + lo] = (__bf16)v;
    }
  }
}

// =====================================================================
// Host-side orchestration
// =====================================================================
extern "C" void kernel_launch(void* const* d_in, const int* in_sizes, int n_in,
                              void* d_out, int out_size, void* d_ws,
                              size_t ws_size, hipStream_t stream) {
  (void)in_sizes; (void)n_in; (void)out_size; (void)ws_size;

  const float* x     = (const float*)d_in[0];
  const float* Wq    = (const float*)d_in[1];
  const float* Wk    = (const float*)d_in[2];
  const float* Wv    = (const float*)d_in[3];
  const float* Wo    = (const float*)d_in[4];
  const float* bo    = (const float*)d_in[5];
  const float* Wspan = (const float*)d_in[6];
  const float* bspan = (const float*)d_in[7];
  float* out = (float*)d_out;

  char* ws = (char*)d_ws;
  const size_t MB = 1024 * 1024;
  __bf16* xb    = (__bf16*)(ws + 0);        //  8 MB  x bf16 (4096x1024)
  __bf16* wqb   = (__bf16*)(ws + 8 * MB);   //  2 MB
  __bf16* wkb   = (__bf16*)(ws + 10 * MB);  //  2 MB
  __bf16* wvb   = (__bf16*)(ws + 12 * MB);  //  2 MB
  __bf16* wob   = (__bf16*)(ws + 14 * MB);  //  2 MB
  __bf16* Qb    = (__bf16*)(ws + 16 * MB);  //  8 MB
  __bf16* Kb    = (__bf16*)(ws + 24 * MB);  //  8 MB
  __bf16* Vtb   = (__bf16*)(ws + 32 * MB);  //  8 MB  V transposed (B, D, T)
  __bf16* Ob    = (__bf16*)(ws + 40 * MB);  //  8 MB  attention output
  float*  xmean = (float*)(ws + 48 * MB);   // 16 KB
  float*  zbuf  = (float*)(ws + 48 * MB + 64 * 1024);  // 256 B

  const int M = cB * cT, N = cD, K = cD;

  cvt_bf16_kernel<<<1024, 256, 0, stream>>>(x, xb, M * K);
  cvt_bf16_kernel<<<512, 256, 0, stream>>>(Wq, wqb, N * K);
  cvt_bf16_kernel<<<512, 256, 0, stream>>>(Wk, wkb, N * K);
  cvt_bf16_kernel<<<512, 256, 0, stream>>>(Wv, wvb, N * K);
  cvt_bf16_kernel<<<512, 256, 0, stream>>>(Wo, wob, N * K);

  dim3 gg(N / 128, M / 128);
  gemm_bf16_kernel<0><<<gg, 256, 0, stream>>>(xb, wqb, Qb, nullptr, nullptr, M, N, K);
  gemm_bf16_kernel<0><<<gg, 256, 0, stream>>>(xb, wkb, Kb, nullptr, nullptr, M, N, K);
  gemm_bf16_kernel<1><<<gg, 256, 0, stream>>>(xb, wvb, Vtb, nullptr, nullptr, M, N, K);

  mean_kernel<<<(cB * cD + 255) / 256, 256, 0, stream>>>(x, xmean);
  span_kernel<<<cB * cH, 256, 0, stream>>>(xmean, Wspan, bspan, zbuf);

  attn_kernel<<<dim3(cT / 128, cB * cH), 256, 0, stream>>>(Qb, Kb, Vtb, zbuf, Ob);

  gemm_bf16_kernel<2><<<gg, 256, 0, stream>>>(Ob, wob, nullptr, out, bo, M, N, K);
}
